// PELSTM_33088428048927
// MI455X (gfx1250) — matrix operand order repaired and checked by reading, not compiled, not run on hardware
//
#include <hip/hip_runtime.h>
#include <hip/hip_bf16.h>

typedef _Float16 half_t;
typedef __attribute__((ext_vector_type(16))) _Float16 v16h;
typedef __attribute__((ext_vector_type(8)))  float    v8f;

namespace {
constexpr int NP_ = 10, XD_ = 6, ZD_ = 64, HD_ = 128, NC_ = 24, TT_ = 1024;
constexpr int DIN_  = XD_ + HD_;   // 134
constexpr int DINP_ = 160;         // padded to 5 K-slabs of 32
constexpr int K0P_  = 224;         // (6+64)+128 padded -> 7 slabs
constexpr int K1P_  = 256;         // 128+128 -> 8 slabs
constexpr int G4_   = 4 * HD_;     // 512

// packed-f16-weight offsets inside workspace (units: halves)
constexpr int W_S1Q = 0, W_S1K = 10240, W_S1V = 20480, W_S1O = 30720;
constexpr int W_S2  = 34816;       // + {0,4096,8192,12288} for q,k,v,o
constexpr int W_STSZ = 51200;      // per set-transformer
constexpr int W_OFC = 102400;      // 5*4*512
constexpr int W_L0  = 112640;      // 7*32*512
constexpr int W_L1  = 227328;      // 8*32*512  (end = 358400 halves = 700 KiB)

// dynamic LDS layout (bytes):
//  hstate[2][24][128] f32 | cstate | ain[3][16][160] f16 |
//  union{ S[2][5][16][64] f32 + att[2][4][16][16] f32  ||  gates[24][512] f32 } |
//  zbuf[16][64] f32 | a0[32][224] f16 | a1[32][256] f16
constexpr int SMEM_BYTES = 2*NC_*HD_*4 + 2*NC_*HD_*4 + 3*16*DINP_*2
                         + 12288*4 + 16*64*4 + 32*K0P_*2 + 32*K1P_*2; // 148480

struct SabP {
  const float *bq, *bk, *bv, *bo, *ln0g, *ln0b, *ln1g, *ln1b;
  int wq, wk, wv, wo;              // packed-weight offsets (halves)
};
struct Params {
  const half_t* wsw;               // packed f16 weights (workspace)
  SabP sab[4];                     // st1.sab1, st1.sab2, st2.sab1, st2.sab2
  const float* ofc_b;
  const float *bih0, *bhh0, *bih1, *bhh1;
  const float *fcW, *fcb;
  const float* x;
  float* out;
  int ofc_w, l0_w, l1_w;
  int Tlen;
};

__device__ __forceinline__ float sigmoidf(float x) { return 1.f / (1.f + __expf(-x)); }

// ---- WMMA fragment loaders (16x16x32 f16, per CDNA5 ISA VGPR layouts) ----
// A (16x32): lane = row (mod 16); lanes<16 carry K k0..k0+7 & k0+16..23,
//            lanes>=16 carry k0+8..15 & k0+24..31.
__device__ __forceinline__ v16h frag_a_f16(const half_t* A, int lda, int row0, int k0, int lane) {
  const half_t* p = A + (row0 + (lane & 15)) * lda + k0 + ((lane & 16) ? 8 : 0);
  v16h f;
#pragma unroll
  for (int e = 0; e < 8; ++e) { f[e] = p[e]; f[8 + e] = p[16 + e]; }
  return f;
}
__device__ __forceinline__ v16h frag_a_f32(const float* A, int lda, int row0, int k0, int lane) {
  const float* p = A + (row0 + (lane & 15)) * lda + k0 + ((lane & 16) ? 8 : 0);
  v16h f;
#pragma unroll
  for (int e = 0; e < 8; ++e) { f[e] = (half_t)p[e]; f[8 + e] = (half_t)p[16 + e]; }
  return f;
}

// One 16x16 output tile-column: C[row0..row0+16, nt*16..+16] over KT K-slabs.
// B is pre-packed in per-lane fragment order: tile(kt,nt) at (nt*KT+kt)*512 halves,
// lane's 16 halves contiguous -> single 32B global load per tile per lane.
__device__ __forceinline__ void tile_mma(bool af16, const void* A, int lda, int row0,
                                         const half_t* wsw, int boff, int KT, int nt,
                                         float* C, int ldc,
                                         const float* b1, const float* b2,
                                         bool relu, int mlimit, int lane)
{
  v8f acc = {};
  const v16h* Bt = (const v16h*)(wsw + boff) + (size_t)nt * KT * 32;
  for (int kt = 0; kt < KT; ++kt) {
    if (kt + 1 < KT) __builtin_prefetch(Bt + (kt + 1) * 32 + lane, 0, 1);
    v16h a = af16 ? frag_a_f16((const half_t*)A, lda, row0, kt * 32, lane)
                  : frag_a_f32((const float*)A, lda, row0, kt * 32, lane);
    v16h bf = Bt[kt * 32 + lane];
    acc = __builtin_amdgcn_wmma_f32_16x16x32_f16(false, a, false, bf, (short)0, acc,
                                                 false, false);
  }
  // C layout: VGPR r -> M=row0+r (lanes 0-15) / M=row0+8+r (lanes 16-31); N=lane&15
  int n  = nt * 16 + (lane & 15);
  int mb = row0 + ((lane & 16) ? 8 : 0);
  float bias = (b1 ? b1[n] : 0.f) + (b2 ? b2[n] : 0.f);
#pragma unroll
  for (int r = 0; r < 8; ++r) {
    int m = mb + r;
    float v = acc[r] + bias;
    if (relu) v = fmaxf(v, 0.f);
    if (m < mlimit) C[m * ldc + n] = v;
  }
}

// ---- prologue: re-tile fp32 weights into f16 WMMA-B fragment order ----
__global__ void pack_kn(const float* __restrict__ W, int K, int N, int KT,
                        half_t* __restrict__ dst)
{
  int idx = blockIdx.x * blockDim.x + threadIdx.x;
  int NT = N >> 4;
  if (idx >= KT * NT * 32) return;
  int l = idx & 31, tt = idx >> 5;
  int kt = tt % KT, nt = tt / KT;
  int n  = nt * 16 + (l & 15);
  int kb = kt * 32 + ((l & 16) ? 8 : 0);
  half_t* d = dst + ((size_t)(nt * KT + kt) << 9) + l * 16;
#pragma unroll
  for (int e = 0; e < 16; ++e) {
    int k = kb + e + (e & 8);
    d[e] = (half_t)((k < K) ? W[(size_t)k * N + n] : 0.f);
  }
}
// fused LSTM B: rows [0,Kih)=W_ih^T, [Kih,Kih+128)=W_hh^T, rest 0 (N=512)
__global__ void pack_lstm(const float* __restrict__ Wih, int Kih,
                          const float* __restrict__ Whh,
                          half_t* __restrict__ dst, int KT)
{
  int idx = blockIdx.x * blockDim.x + threadIdx.x;
  if (idx >= KT * 32 * 32) return;
  int l = idx & 31, tt = idx >> 5;
  int kt = tt % KT, nt = tt / KT;
  int n  = nt * 16 + (l & 15);
  int kb = kt * 32 + ((l & 16) ? 8 : 0);
  half_t* d = dst + ((size_t)(nt * KT + kt) << 9) + l * 16;
#pragma unroll
  for (int e = 0; e < 16; ++e) {
    int k = kb + e + (e & 8);
    float v = 0.f;
    if (k < Kih)            v = Wih[(size_t)n * Kih + k];
    else if (k < Kih + HD_) v = Whh[(size_t)n * HD_ + (k - Kih)];
    d[e] = (half_t)v;
  }
}

// ---- VALU attention (4 heads, 10x10, dh=16) + LN(q+o) ----
__device__ void attn_ln(const Params& P, int stage, int qb, int kb, int vb, int ob, int hb,
                        float* S, float* att, int tid)
{
  for (int i = tid; i < 800; i += 256) {            // scores, scale = 1/sqrt(64)
    int tm = i / 400, rem = i - tm * 400;
    int h = rem / 100, ij = rem - h * 100;
    int r = ij / 10, c = ij - r * 10;
    const float* q = S + (tm * 5 + qb) * 1024;
    const float* k = S + (tm * 5 + kb) * 1024;
    float s = 0.f;
#pragma unroll
    for (int d = 0; d < 16; ++d) s += q[r * 64 + h * 16 + d] * k[c * 64 + h * 16 + d];
    att[((tm * 4 + h) * 16 + r) * 16 + c] = s * 0.125f;
  }
  __syncthreads();
  for (int i = tid; i < 80; i += 256) {             // softmax rows
    int tm = i / 40, rem = i - tm * 40;
    int h = rem / 10, r = rem - h * 10;
    float* a = att + ((tm * 4 + h) * 16 + r) * 16;
    float mx = a[0];
#pragma unroll
    for (int c = 1; c < 10; ++c) mx = fmaxf(mx, a[c]);
    float sum = 0.f;
#pragma unroll
    for (int c = 0; c < 10; ++c) { float e = __expf(a[c] - mx); a[c] = e; sum += e; }
    float inv = 1.f / sum;
#pragma unroll
    for (int c = 0; c < 10; ++c) a[c] *= inv;
  }
  __syncthreads();
  for (int i = tid; i < 1280; i += 256) {           // o = att @ v
    int tm = i / 640, rem = i - tm * 640;
    int r = rem / 64, col = rem - r * 64;
    int h = col >> 4;
    const float* v = S + (tm * 5 + vb) * 1024;
    const float* a = att + ((tm * 4 + h) * 16 + r) * 16;
    float s = 0.f;
#pragma unroll
    for (int j = 0; j < 10; ++j) s += a[j] * v[j * 64 + col];
    S[(tm * 5 + ob) * 1024 + r * 64 + col] = s;
  }
  __syncthreads();
  for (int i = tid; i < 20; i += 256) {             // h = LN(q + o)
    int tm = i / 10, r = i - tm * 10;
    const SabP& sp = P.sab[tm * 2 + stage];
    const float* q = S + (tm * 5 + qb) * 1024 + r * 64;
    const float* o = S + (tm * 5 + ob) * 1024 + r * 64;
    float* dst = S + (tm * 5 + hb) * 1024 + r * 64;
    float m = 0.f;
    for (int c = 0; c < 64; ++c) m += q[c] + o[c];
    m *= (1.f / 64.f);
    float var = 0.f;
    for (int c = 0; c < 64; ++c) { float d = q[c] + o[c] - m; var += d * d; }
    var *= (1.f / 64.f);
    float inv = rsqrtf(var + 1e-5f);
    for (int c = 0; c < 64; ++c)
      dst[c] = (q[c] + o[c] - m) * inv * sp.ln0g[c] + sp.ln0b[c];
  }
  __syncthreads();
}

__device__ void ln2(const Params& P, int stage, int hb, int fb, int zb, float* S, int tid)
{
  for (int i = tid; i < 20; i += 256) {             // z = LN(h + ff)
    int tm = i / 10, r = i - tm * 10;
    const SabP& sp = P.sab[tm * 2 + stage];
    const float* h = S + (tm * 5 + hb) * 1024 + r * 64;
    const float* f = S + (tm * 5 + fb) * 1024 + r * 64;
    float* dst = S + (tm * 5 + zb) * 1024 + r * 64;
    float m = 0.f;
    for (int c = 0; c < 64; ++c) m += h[c] + f[c];
    m *= (1.f / 64.f);
    float var = 0.f;
    for (int c = 0; c < 64; ++c) { float d = h[c] + f[c] - m; var += d * d; }
    var *= (1.f / 64.f);
    float inv = rsqrtf(var + 1e-5f);
    for (int c = 0; c < 64; ++c)
      dst[c] = (h[c] + f[c] - m) * inv * sp.ln1g[c] + sp.ln1b[c];
  }
  __syncthreads();
}

__device__ void sab_post(const Params& P, int stage, int qb, int kb, int vb, int ob,
                         int hb, int fb, int zb, float* S, float* att,
                         int tid, int wave, int lane)
{
  attn_ln(P, stage, qb, kb, vb, ob, hb, S, att, tid);
  {                                                  // ff = relu(h @ Wo + bo): 8 jobs
    int j = wave, tm = j >> 2, nt = j & 3;
    const SabP& sp = P.sab[tm * 2 + stage];
    tile_mma(false, S + (tm * 5 + hb) * 1024, 64, 0, P.wsw, sp.wo, 2, nt,
             S + (tm * 5 + fb) * 1024, 64, sp.bo, nullptr, true, 16, lane);
  }
  __syncthreads();
  ln2(P, stage, hb, fb, zb, S, tid);
}

// ---- persistent kernel: one block per batch element, loop over T ----
__global__ void __launch_bounds__(256) pelstm_main(Params P)
{
  extern __shared__ char smem[];
  float*  hst  = (float*)smem;                       // [2][24][128]
  float*  cst  = hst + 2 * NC_ * HD_;
  half_t* ain  = (half_t*)(cst + 2 * NC_ * HD_);     // [3][16][160]
  float*  uni  = (float*)(ain + 3 * 16 * DINP_);
  float*  S    = uni;                                // [2][5][16][64]
  float*  att  = uni + 2 * 5 * 16 * 64;              // [2][4][16][16]
  float*  gates= uni;                                // [24][512]  (aliases S+att)
  float*  zbuf = uni + 12288;                        // [16][64]
  half_t* a0   = (half_t*)(zbuf + 16 * 64);          // [32][224]
  half_t* a1   = a0 + 32 * K0P_;                     // [32][256]

  const int b = blockIdx.x;
  const int tid = threadIdx.x, lane = tid & 31, wave = tid >> 5;
  float* hst0 = hst;              float* hst1 = hst + NC_ * HD_;
  float* cst0 = cst;              float* cst1 = cst + NC_ * HD_;

  for (int i = tid; i < 2 * NC_ * HD_; i += 256) { hst[i] = 0.f; cst[i] = 0.f; }
  __syncthreads();

  for (int t = 0; t < P.Tlen; ++t) {
    const float* xp = P.x + ((size_t)b * P.Tlen + t) * (XD_ * NC_);

    // P0: build SAB inputs: [x | h_last | 0-pad], teams 0/1 (10 rows), outside (4 rows)
    for (int i = tid; i < 3 * 16 * DINP_; i += 256) {
      int buf = i / (16 * DINP_), rem = i - buf * 16 * DINP_;
      int r = rem / DINP_, c = rem - r * DINP_;
      float v = 0.f;
      int nrow = (buf == 2) ? 4 : NP_;
      if (r < nrow) {
        int gr = buf * NP_ + r;                      // 0..9 / 10..19 / 20..23
        if (c < XD_)       v = xp[gr * XD_ + c];
        else if (c < DIN_) v = hst1[gr * HD_ + (c - XD_)];
      }
      ain[i] = (half_t)v;
    }
    __syncthreads();

    // P1: SAB1 Q/K/V (both teams) + outside FC   (28 WMMA tile-columns, KT=5)
    for (int j = wave; j < 28; j += 8) {
      int mat = j >> 2, nt = j & 3;
      if (mat < 6) {
        int tm = mat / 3, which = mat - tm * 3;
        const SabP& sp = P.sab[tm * 2];
        int boff = (which == 0) ? sp.wq : (which == 1) ? sp.wk : sp.wv;
        const float* bias = (which == 0) ? sp.bq : (which == 1) ? sp.bk : sp.bv;
        tile_mma(true, ain + tm * 16 * DINP_, DINP_, 0, P.wsw, boff, 5, nt,
                 S + (tm * 5 + which) * 1024, 64, bias, nullptr, false, 16, lane);
      } else {
        tile_mma(true, ain + 2 * 16 * DINP_, DINP_, 0, P.wsw, P.ofc_w, 5, nt,
                 zbuf, 64, P.ofc_b, nullptr, true, 16, lane);
      }
    }
    __syncthreads();

    // SAB1 attention + LN + rFF + LN   (q=0,k=1,v=2,o=3 -> h=1, ff=2, z=0)
    sab_post(P, 0, 0, 1, 2, 3, 1, 2, 0, S, att, tid, wave, lane);

    // SAB2 Q/K/V from z (buf0), KT=2   (24 tile-columns)
    for (int j = wave; j < 24; j += 8) {
      int tm = j / 12, rem = j - tm * 12;
      int which = rem >> 2, nt = rem & 3;
      const SabP& sp = P.sab[tm * 2 + 1];
      int boff = (which == 0) ? sp.wq : (which == 1) ? sp.wk : sp.wv;
      const float* bias = (which == 0) ? sp.bq : (which == 1) ? sp.bk : sp.bv;
      tile_mma(false, S + (tm * 5 + 0) * 1024, 64, 0, P.wsw, boff, 2, nt,
               S + (tm * 5 + 1 + which) * 1024, 64, bias, nullptr, false, 16, lane);
    }
    __syncthreads();

    // SAB2 tail  (q=1,k=2,v=3,o=4 -> h=0, ff=2, z=1)
    sab_post(P, 1, 1, 2, 3, 4, 0, 2, 1, S, att, tid, wave, lane);

    // P15: LSTM layer-0 input  [x_t(6) | z_t(64) | h0_prev(128) | 0-pad]
    for (int i = tid; i < 32 * K0P_; i += 256) {
      int r = i / K0P_, c = i - r * K0P_;
      float v = 0.f;
      if (r < NC_) {
        if (c < XD_) v = xp[r * XD_ + c];
        else if (c < XD_ + ZD_) {
          int cc = c - XD_;
          if (r < NP_)          v = S[(0 * 5 + 1) * 1024 + r * 64 + cc];
          else if (r < 2 * NP_) v = S[(1 * 5 + 1) * 1024 + (r - NP_) * 64 + cc];
          else                  v = zbuf[(r - 2 * NP_) * 64 + cc];
        } else if (c < XD_ + ZD_ + HD_) {
          v = hst0[r * HD_ + (c - (XD_ + ZD_))];
        }
      }
      a0[i] = (half_t)v;
    }
    __syncthreads();

    // P16: layer-0 gates = A0 @ [Wih0;Whh0]^pk + bih0 + bhh0   (64 tiles, KT=7)
    for (int j = wave; j < 64; j += 8) {
      int mt = j >> 5, nt = j & 31;
      tile_mma(true, a0, K0P_, mt * 16, P.wsw, P.l0_w, 7, nt,
               gates, G4_, P.bih0, P.bhh0, false, NC_, lane);
    }
    __syncthreads();

    // P17: cell layer 0 (torch gate order i,f,g,o) + build layer-1 input
    for (int i = tid; i < NC_ * HD_; i += 256) {
      int r = i >> 7, d = i & 127;
      float gi = gates[r * G4_ + d];
      float gf = gates[r * G4_ + HD_ + d];
      float gg = gates[r * G4_ + 2 * HD_ + d];
      float go = gates[r * G4_ + 3 * HD_ + d];
      float c2 = sigmoidf(gf) * cst0[i] + sigmoidf(gi) * tanhf(gg);
      float h2 = sigmoidf(go) * tanhf(c2);
      cst0[i] = c2; hst0[i] = h2;
      a1[r * K1P_ + d] = (half_t)h2;
    }
    for (int i = tid; i < 32 * HD_; i += 256) {
      int r = i >> 7, d = i & 127;
      float v = (r < NC_) ? hst1[r * HD_ + d] : 0.f;
      a1[r * K1P_ + HD_ + d] = (half_t)v;
      if (r >= NC_) a1[r * K1P_ + d] = (half_t)0.f;
    }
    __syncthreads();

    // P18: layer-1 gates (64 tiles, KT=8)
    for (int j = wave; j < 64; j += 8) {
      int mt = j >> 5, nt = j & 31;
      tile_mma(true, a1, K1P_, mt * 16, P.wsw, P.l1_w, 8, nt,
               gates, G4_, P.bih1, P.bhh1, false, NC_, lane);
    }
    __syncthreads();

    // P19: cell layer 1
    for (int i = tid; i < NC_ * HD_; i += 256) {
      int r = i >> 7, d = i & 127;
      float gi = gates[r * G4_ + d];
      float gf = gates[r * G4_ + HD_ + d];
      float gg = gates[r * G4_ + 2 * HD_ + d];
      float go = gates[r * G4_ + 3 * HD_ + d];
      float c2 = sigmoidf(gf) * cst1[i] + sigmoidf(gi) * tanhf(gg);
      float h2 = sigmoidf(go) * tanhf(c2);
      cst1[i] = c2; hst1[i] = h2;
    }
    __syncthreads();

    // P20: GLU head: pred = lin0 * sigmoid(lin1)
    for (int r = tid; r < NC_; r += 256) {
      float l0 = P.fcb[0], l1 = P.fcb[1];
      for (int d = 0; d < HD_; ++d) {
        float h = hst1[r * HD_ + d];
        l0 += h * P.fcW[2 * d + 0];
        l1 += h * P.fcW[2 * d + 1];
      }
      P.out[((size_t)b * P.Tlen + t) * NC_ + r] = l0 * sigmoidf(l1);
    }
    __syncthreads();
  }
}
} // namespace

extern "C" void kernel_launch(void* const* d_in, const int* in_sizes, int n_in,
                              void* d_out, int out_size, void* d_ws, size_t ws_size,
                              hipStream_t stream)
{
  (void)out_size; (void)ws_size;
  auto F = [&](int i) { return (const float*)d_in[i]; };

  // Leaf order: default = JAX pytree (sorted dict keys). If leaf 0 is the huge
  // x tensor, the harness used insertion order instead.
  bool sorted_order = !(n_in > 0 && in_sizes[0] > 1000000);

  const float *fcW, *fcb, *Whh0, *Whh1, *Wih0, *Wih1, *bhh0, *bhh1, *bih0, *bih1,
              *ofcW, *ofcb, *x;
  const float *sWq[4], *sWk[4], *sWv[4], *sWo[4], *sbq[4], *sbk[4], *sbv[4], *sbo[4],
              *sl0g[4], *sl0b[4], *sl1g[4], *sl1b[4];
  int xIdx;
  if (sorted_order) {
    // fc{W,b}, lstm{W_hh0,W_hh1,W_ih0,W_ih1,b_hh0,b_hh1,b_ih0,b_ih1}, out_fc{W,b},
    // st1{sab1,sab2}, st2{sab1,sab2} each: Wk,Wo,Wq,Wv,bk,bo,bq,bv,ln0_b,ln0_g,ln1_b,ln1_g, x
    fcW = F(0); fcb = F(1); Whh0 = F(2); Whh1 = F(3); Wih0 = F(4); Wih1 = F(5);
    bhh0 = F(6); bhh1 = F(7); bih0 = F(8); bih1 = F(9); ofcW = F(10); ofcb = F(11);
    for (int s4 = 0; s4 < 4; ++s4) {
      int bi = 12 + s4 * 12;
      sWk[s4] = F(bi + 0); sWo[s4] = F(bi + 1); sWq[s4] = F(bi + 2); sWv[s4] = F(bi + 3);
      sbk[s4] = F(bi + 4); sbo[s4] = F(bi + 5); sbq[s4] = F(bi + 6); sbv[s4] = F(bi + 7);
      sl0b[s4] = F(bi + 8); sl0g[s4] = F(bi + 9); sl1b[s4] = F(bi + 10); sl1g[s4] = F(bi + 11);
    }
    x = F(60); xIdx = 60;
  } else {
    x = F(0); xIdx = 0;
    for (int s4 = 0; s4 < 4; ++s4) {
      int bi = 1 + s4 * 12;  // Wq,bq,Wk,bk,Wv,bv,Wo,bo,ln0_g,ln0_b,ln1_g,ln1_b
      sWq[s4] = F(bi + 0); sbq[s4] = F(bi + 1); sWk[s4] = F(bi + 2); sbk[s4] = F(bi + 3);
      sWv[s4] = F(bi + 4); sbv[s4] = F(bi + 5); sWo[s4] = F(bi + 6); sbo[s4] = F(bi + 7);
      sl0g[s4] = F(bi + 8); sl0b[s4] = F(bi + 9); sl1g[s4] = F(bi + 10); sl1b[s4] = F(bi + 11);
    }
    ofcW = F(49); ofcb = F(50);
    Wih0 = F(51); Wih1 = F(52); Whh0 = F(53); Whh1 = F(54);
    bih0 = F(55); bih1 = F(56); bhh0 = F(57); bhh1 = F(58);
    fcW = F(59); fcb = F(60);
  }

  half_t* wsw = (half_t*)d_ws;
  auto pkn = [&](const float* W, int K, int Nn, int KT, int off) {
    int total = KT * (Nn / 16) * 32;
    pack_kn<<<(total + 255) / 256, 256, 0, stream>>>(W, K, Nn, KT, wsw + off);
  };
  for (int st = 0; st < 2; ++st) {
    int wb = st * W_STSZ;
    pkn(sWq[st * 2], DIN_, 64, 5, wb + W_S1Q);
    pkn(sWk[st * 2], DIN_, 64, 5, wb + W_S1K);
    pkn(sWv[st * 2], DIN_, 64, 5, wb + W_S1V);
    pkn(sWo[st * 2], 64, 64, 2, wb + W_S1O);
    pkn(sWq[st * 2 + 1], 64, 64, 2, wb + W_S2 + 0);
    pkn(sWk[st * 2 + 1], 64, 64, 2, wb + W_S2 + 4096);
    pkn(sWv[st * 2 + 1], 64, 64, 2, wb + W_S2 + 8192);
    pkn(sWo[st * 2 + 1], 64, 64, 2, wb + W_S2 + 12288);
  }
  pkn(ofcW, DIN_, 64, 5, W_OFC);
  pack_lstm<<<(7 * 32 * 32 + 255) / 256, 256, 0, stream>>>(Wih0, 70, Whh0, wsw + W_L0, 7);
  pack_lstm<<<(8 * 32 * 32 + 255) / 256, 256, 0, stream>>>(Wih1, HD_, Whh1, wsw + W_L1, 8);

  Params P{};
  P.wsw = wsw;
  for (int s4 = 0; s4 < 4; ++s4) {
    SabP& sp = P.sab[s4];
    sp.bq = sbq[s4]; sp.bk = sbk[s4]; sp.bv = sbv[s4]; sp.bo = sbo[s4];
    sp.ln0g = sl0g[s4]; sp.ln0b = sl0b[s4]; sp.ln1g = sl1g[s4]; sp.ln1b = sl1b[s4];
    int wb = (s4 >> 1) * W_STSZ;
    if ((s4 & 1) == 0) { sp.wq = wb + W_S1Q; sp.wk = wb + W_S1K; sp.wv = wb + W_S1V; sp.wo = wb + W_S1O; }
    else { sp.wq = wb + W_S2; sp.wk = wb + W_S2 + 4096; sp.wv = wb + W_S2 + 8192; sp.wo = wb + W_S2 + 12288; }
  }
  P.ofc_b = ofcb; P.ofc_w = W_OFC;
  P.bih0 = bih0; P.bhh0 = bhh0; P.bih1 = bih1; P.bhh1 = bhh1;
  P.fcW = fcW; P.fcb = fcb;
  P.x = x; P.out = (float*)d_out;
  P.l0_w = W_L0; P.l1_w = W_L1;
  P.Tlen = TT_;

  int Bc = 128;
  if (xIdx < n_in && in_sizes[xIdx] > 0) {
    int cand = in_sizes[xIdx] / (TT_ * XD_ * NC_);
    if (cand > 0) Bc = cand;
  }
  pelstm_main<<<Bc, 256, SMEM_BYTES, stream>>>(P);
}
